// audioDecoder_29472065585241
// MI455X (gfx1250) — matrix operand order repaired and checked by reading, not compiled, not run on hardware
//
#include <hip/hip_runtime.h>
#include <hip/hip_bf16.h>

typedef __attribute__((ext_vector_type(16))) _Float16 v16h;
typedef __attribute__((ext_vector_type(8)))  _Float16 v8h;
typedef __attribute__((ext_vector_type(4)))  _Float16 v4h;
typedef __attribute__((ext_vector_type(8)))  float    v8f;

#define N_CH 256          // encoder channels
#define B_CH 128          // bottleneck channels
#define KFR  32000        // frames
#define TOUT 256008       // output samples per batch (pad == 0)
#define TN   128          // frames per tile
#define NBLK (KFR / TN)   // 250 tiles per batch
#define NBW  50           // blocks per batch
#define TPB  (NBLK / NBW) // 5 tiles per block

union V16U { v16h v; v8h h[2]; };
__device__ __forceinline__ v16h join8(v8h lo, v8h hi) { V16U u; u.h[0] = lo; u.h[1] = hi; return u.v; }
__device__ __forceinline__ v4h cvt4(float4 v) {
    v4h r; r[0] = (_Float16)v.x; r[1] = (_Float16)v.y; r[2] = (_Float16)v.z; r[3] = (_Float16)v.w; return r;
}

__global__ void __launch_bounds__(256) zero_f32(float* __restrict__ p, int n4) {
    int i = blockIdx.x * blockDim.x + threadIdx.x;
    if (i < n4) ((float4*)p)[i] = float4{0.f, 0.f, 0.f, 0.f};
}

__global__ void __launch_bounds__(256) fused_decoder(
    const float* __restrict__ mix,   // [8, 256, 32000]
    const float* __restrict__ est,   // [8, 128, 32000]
    const float* __restrict__ Wm,    // [256, 128]
    const float* __restrict__ Wb,    // [16, 256]
    float* __restrict__ out)         // [8, 256008]
{
    __shared__ _Float16 sWm[N_CH][B_CH];    // 64 KB  W_mask (f16), lives all kernel
    __shared__ _Float16 sE [B_CH][TN];      // 32 KB  est_mask tile (f16)
    __shared__ _Float16 sXs[8][32][16];     //  8 KB  per-wave 32ch x 16fr x-chunk ping buffer
                                            // total 104 KB -> 3 workgroups / WGP

    const int tid  = threadIdx.x;
    const int wave = tid >> 5;
    const int lane = tid & 31;
    const int lh   = lane & 15;
    const int hi   = lane >> 4;              // 0 or 1
    const int g    = blockIdx.x / NBW;       // batch
    const int t0   = (blockIdx.x - g * NBW) * TPB;   // first tile index

    // ---- W_mask -> LDS as f16 (once per block, one 128-float row per thread) ----
    {
        const float4* src = (const float4*)(Wm + (size_t)tid * B_CH);
        #pragma unroll
        for (int i = 0; i < B_CH / 4; ++i)
            *(v4h*)&sWm[tid][4 * i] = cvt4(src[i]);
    }
    // ---- W_basis -> resident A operands (8 chunks of K=32, reused for all tiles) ----
    // A layout: lane%16 = M row; element e <-> K = (e<8 ? e : e+8) + 8*hi
    v16h wb[8];
    {
        const float* wrow = Wb + (size_t)lh * N_CH;   // row l = lane%16
        #pragma unroll
        for (int s = 0; s < 8; ++s)
            #pragma unroll
            for (int e = 0; e < 16; ++e) {
                int k = 32 * s + ((e < 8) ? e : e + 8) + 8 * hi;
                wb[s][e] = (_Float16)wrow[k];
            }
    }

    const int wf = wave * 16;                // this wave's 16 frames within a tile

    for (int t = 0; t < TPB; ++t) {
        const int fb = (t0 + t) * TN;        // frame tile base
        const int gf = fb + wf;              // global frame base for this wave

        // ---- est_mask tile -> LDS as f16 (coalesced, half a row per thread) ----
        {
            const int ch = tid >> 1, half = tid & 1;
            const float4* src = (const float4*)(est + ((size_t)g * B_CH + ch) * KFR + fb + 64 * half);
            #pragma unroll
            for (int i = 0; i < 16; ++i)
                *(v4h*)&sE[ch][64 * half + 4 * i] = cvt4(src[i]);
        }
        __syncthreads();

        // ---- B operands for matmul1 (hoisted; reused across all 16 M-tiles) ----
        // B layout: lane%16 = N frame (wf + lh); element e <-> K = 32c + (e<8 ? e : e+8) + 8*hi
        v16h eb[4];
        #pragma unroll
        for (int c = 0; c < 4; ++c) {
            #pragma unroll
            for (int e = 0; e < 16; ++e)
                eb[c][e] = sE[32 * c + ((e < 8) ? e : e + 8) + 8 * hi][wf + lh];
        }
        __syncthreads();   // sE consumed into registers; free for next tile

        // ---- fused matmul1 (relu * mixture) feeding matmul2 chunk-by-chunk ----
        const float* mbase = mix + (size_t)g * N_CH * KFR + gf + lh;
        float mv[8], nv[8];
        #pragma unroll
        for (int j = 0; j < 8; ++j)          // prefetch mixture rows for mt = 0
            mv[j] = mbase[(size_t)(8 * hi + j) * KFR];

        v8f acc2 = {};
        #pragma unroll
        for (int s = 0; s < 8; ++s) {        // 32-channel chunk = M-tile pair (2s, 2s+1)
            #pragma unroll
            for (int half = 0; half < 2; ++half) {
                const int mt = 2 * s + half;
                if (mt < 15) {               // software-pipeline next M-tile's mixture loads
                    const float* p = mbase + (size_t)(16 * (mt + 1) + 8 * hi) * KFR;
                    #pragma unroll
                    for (int j = 0; j < 8; ++j) nv[j] = p[(size_t)j * KFR];
                }
                // batch all 4 A-chunks (8x ds_load_b128 in one clause), then WMMA
                // with staggered dscnt waits instead of a full wait per chunk
                const int row = 16 * mt + lh;
                v16h a[4];
                #pragma unroll
                for (int c = 0; c < 4; ++c) {
                    const v8h* p0 = (const v8h*)&sWm[row][32 * c + 8 * hi];
                    const v8h* p1 = (const v8h*)&sWm[row][32 * c + 16 + 8 * hi];
                    a[c] = join8(p0[0], p1[0]);
                }
                v8f acc = {};
                #pragma unroll
                for (int c = 0; c < 4; ++c)
                    acc = __builtin_amdgcn_wmma_f32_16x16x32_f16(false, a[c], false, eb[c],
                                                                 (short)0, acc, false, false);
                // C layout: VGPR j -> channel 16mt + j + 8hi, frame gf + lh
                #pragma unroll
                for (int j = 0; j < 8; ++j) {
                    float v  = acc[j];
                    float xv = mv[j] * (v > 0.f ? v : 0.f);
                    sXs[wave][16 * half + 8 * hi + j][lh] = (_Float16)xv;
                }
                if (mt < 15) {
                    #pragma unroll
                    for (int j = 0; j < 8; ++j) mv[j] = nv[j];
                }
            }
            // matmul2 accumulate: B = x chunk (32 ch x 16 fr), wave-private LDS
            v16h xb;
            #pragma unroll
            for (int e = 0; e < 16; ++e)
                xb[e] = sXs[wave][((e < 8) ? e : e + 8) + 8 * hi][lh];
            acc2 = __builtin_amdgcn_wmma_f32_16x16x32_f16(false, wb[s], false, xb,
                                                          (short)0, acc2, false, false);
        }

        // ---- overlap-add (hop 8, len 16): out[g, 8*frame + l] += x2[frame, l] ----
        float* obase = out + (size_t)g * TOUT;
        const int fr = gf + lh;
        #pragma unroll
        for (int j = 0; j < 8; ++j) {
            int l = j + 8 * hi;
            unsafeAtomicAdd(&obase[8 * fr + l], acc2[j]);   // global_atomic_add_f32
        }
    }
}

extern "C" void kernel_launch(void* const* d_in, const int* in_sizes, int n_in,
                              void* d_out, int out_size, void* d_ws, size_t ws_size,
                              hipStream_t stream) {
    const float* mix = (const float*)d_in[0];   // mixture_w  [8,256,32000]
    const float* est = (const float*)d_in[1];   // est_mask   [8,128,32000]
    const float* Wm  = (const float*)d_in[2];   // W_mask     [256,128]
    const float* Wb  = (const float*)d_in[3];   // W_basis    [16,256]
    float* out = (float*)d_out;                 // [8,256008]

    const int n4 = out_size / 4;
    zero_f32<<<(n4 + 255) / 256, 256, 0, stream>>>(out, n4);
    fused_decoder<<<8 * NBW, 256, 0, stream>>>(mix, est, Wm, Wb, out);
}
